// CTCDecodingLayer_6296422055982
// MI455X (gfx1250) — compile-verified, matrix operand order checked by reading, not run on hardware
//
#include <hip/hip_runtime.h>
#include <hip/hip_bf16.h>

// CTC greedy decode for B=256, T=2048, C=80 (blank = 79).
// Phase 1: wave-per-timestep argmax (bandwidth bound, ~168MB streamed).
// Phase 2: wave-per-row scan via v_wmma_f32_16x16x32_f16 + scatter.

#define CTC_B 256
#define CTC_T 2048
#define CTC_C 80
#define CTC_BLANK (CTC_C - 1)

typedef __attribute__((ext_vector_type(16))) _Float16 v16h;
typedef __attribute__((ext_vector_type(8)))  float    v8f;

// ---------------------------------------------------------------------------
// Phase 1: argmax over C=80 per (b,t). One wave32 per timestep.
// ---------------------------------------------------------------------------
__device__ __forceinline__ unsigned mono_bits(float f) {
    // Order-preserving float -> uint mapping (works for +/- values).
    unsigned u = __float_as_uint(f);
    return (u & 0x80000000u) ? ~u : (u | 0x80000000u);
}

__global__ void ctc_argmax_kernel(const float* __restrict__ y,
                                  int* __restrict__ ids) {
    const int lane = threadIdx.x & 31;
    const int wave = threadIdx.x >> 5;
    const long long i = (long long)blockIdx.x * 8 + wave;   // 0 .. B*T-1
    const float* p = y + i * CTC_C;

    // 80 channels: lanes 0..31 -> c=lane, c=lane+32; lanes 0..15 -> c=lane+64.
    float v0 = p[lane];
    float v1 = p[lane + 32];
    float v2 = (lane < 16) ? p[lane + 64] : __uint_as_float(0xff800000u); // -inf

    // key = (monotonic bits << 32) | (255 - idx): max-reduce picks the largest
    // value; on exact ties the SMALLEST index wins (matches jnp.argmax).
    unsigned long long k0 = ((unsigned long long)mono_bits(v0) << 32) |
                            (unsigned)(255 - lane);
    unsigned long long k1 = ((unsigned long long)mono_bits(v1) << 32) |
                            (unsigned)(255 - (lane + 32));
    unsigned long long k2 = ((unsigned long long)mono_bits(v2) << 32) |
                            (unsigned)(255 - (lane + 64));
    unsigned long long best = k0 > k1 ? k0 : k1;
    best = best > k2 ? best : k2;

    #pragma unroll
    for (int off = 16; off >= 1; off >>= 1) {
        unsigned long long o = __shfl_xor(best, off, 32);
        best = best > o ? best : o;
    }
    if (lane == 0) {
        ids[i] = 255 - (int)(best & 0xFFu);
    }
}

// ---------------------------------------------------------------------------
// Phase 2: per-row keep mask -> inclusive prefix sum via WMMA -> scatter.
// One wave32 per row (blockDim = 32, gridDim = B).
//
// WMMA formulation: flags of 16 chunks x 32 elems form A (16x32 f16).
// B1[k][n] = (k <= n),  B2[k][n] = (k <= n+16)  (constant triangular ones).
// D1[m][n] = prefix of chunk m at position n; D2[m][n] = at position n+16.
// Chunk totals = D2[m][15]; 64 chunk offsets scanned serially (tiny).
// ---------------------------------------------------------------------------

// ISA 7.12.2 packing: for A(16x32 f16) and B(32x16 f16), lane half selects the
// K-subgroup; half h maps to K = (h<8 ? h : h+8) + (lane>=16 ? 8 : 0).
__device__ __forceinline__ int k_of(int lane, int h) {
    return ((h < 8) ? h : h + 8) + ((lane >= 16) ? 8 : 0);
}

__global__ void ctc_scan_scatter_kernel(const int* __restrict__ ids,
                                        int* __restrict__ out) {
    const int row  = blockIdx.x;
    const int lane = threadIdx.x;            // 0..31, single wave
    const int base = row * CTC_T;

    __shared__ _Float16 s_cum[CTC_T];        // flags, later overwritten by cums
    __shared__ int      s_ids[CTC_T];
    __shared__ float    s_chunk_total[CTC_T / 32];
    __shared__ float    s_chunk_off[CTC_T / 32];

    // Load ids, init output row to -1.
    for (int j = lane; j < CTC_T; j += 32) {
        s_ids[j] = ids[base + j];
        out[base + j] = -1;
    }
    __syncthreads();

    // keep[t] = (id != blank) && (id != id[t-1]); keep[0] vs prev = -1.
    for (int j = lane; j < CTC_T; j += 32) {
        int id   = s_ids[j];
        int prev = (j == 0) ? -1 : s_ids[j - 1];
        bool keep = (id != CTC_BLANK) && (id != prev);
        s_cum[j] = keep ? (_Float16)1.0f : (_Float16)0.0f;
    }
    __syncthreads();

    // Constant triangular B matrices (N = lane mod 16).
    const int nB = lane & 15;
    v16h B1, B2;
    #pragma unroll
    for (int h = 0; h < 16; ++h) {
        int k = k_of(lane, h);
        B1[h] = (k <= nB)      ? (_Float16)1.0f : (_Float16)0.0f;
        B2[h] = (k <= nB + 16) ? (_Float16)1.0f : (_Float16)0.0f;
    }

    const int mA = lane & 15;                 // A row held by this lane
    #pragma unroll
    for (int tile = 0; tile < CTC_T / 512; ++tile) {
        const int tbase = tile * 512;

        // Build A: A[m][k] = flag[tbase + m*32 + k].
        v16h A;
        #pragma unroll
        for (int h = 0; h < 16; ++h) {
            A[h] = s_cum[tbase + mA * 32 + k_of(lane, h)];
        }

        v8f c = {};
        v8f D1 = __builtin_amdgcn_wmma_f32_16x16x32_f16(
                     false, A, false, B1, (short)0, c, false, false);
        v8f D2 = __builtin_amdgcn_wmma_f32_16x16x32_f16(
                     false, A, false, B2, (short)0, c, false, false);

        __syncthreads();   // all flag reads done before overwriting with cums

        // D layout: reg r -> M = r (lanes 0-15) or r+8 (lanes 16-31); N = lane&15.
        const int mOfs = (lane < 16) ? 0 : 8;
        const int n    = lane & 15;
        #pragma unroll
        for (int r = 0; r < 8; ++r) {
            int mm = r + mOfs;
            int t0 = tbase + mm * 32 + n;
            s_cum[t0]      = (_Float16)D1[r];   // prefix at pos n
            s_cum[t0 + 16] = (_Float16)D2[r];   // prefix at pos n+16
            if (n == 15) s_chunk_total[tile * 16 + mm] = D2[r];
        }
        __syncthreads();
    }

    // Serial scan of 64 chunk totals (tiny).
    if (lane == 0) {
        float acc = 0.0f;
        for (int i = 0; i < CTC_T / 32; ++i) {
            s_chunk_off[i] = acc;
            acc += s_chunk_total[i];
        }
    }
    __syncthreads();

    // Scatter: inclusive cum -> slot = cum-1 (keep recomputed from ids).
    for (int j = lane; j < CTC_T; j += 32) {
        int id   = s_ids[j];
        int prev = (j == 0) ? -1 : s_ids[j - 1];
        bool keep = (id != CTC_BLANK) && (id != prev);
        if (keep) {
            int cum = (int)(float)s_cum[j] + (int)s_chunk_off[j >> 5];
            out[base + cum - 1] = id;
        }
    }
}

// ---------------------------------------------------------------------------
extern "C" void kernel_launch(void* const* d_in, const int* in_sizes, int n_in,
                              void* d_out, int out_size, void* d_ws, size_t ws_size,
                              hipStream_t stream) {
    (void)in_sizes; (void)n_in; (void)out_size; (void)ws_size;
    const float* y_pred = (const float*)d_in[0];
    int* out = (int*)d_out;          // int32 output per reference
    int* ids = (int*)d_ws;           // B*T ints = 2 MB scratch

    // Phase 1: 8 waves/block, one wave per timestep.
    const int total = CTC_B * CTC_T;                 // 524288
    ctc_argmax_kernel<<<total / 8, 256, 0, stream>>>(y_pred, ids);

    // Phase 2: one wave per row.
    ctc_scan_scatter_kernel<<<CTC_B, 32, 0, stream>>>(ids, out);
}
